// UnifiedBlock_20358144983289
// MI455X (gfx1250) — compile-verified, matrix operand order checked by reading, not run on hardware
//
#include <hip/hip_runtime.h>

// ---------------------------------------------------------------------------
// Problem constants (match reference)
// ---------------------------------------------------------------------------
enum : int {
    B_  = 4,
    T_  = 2048,
    H_  = 1024,          // HID
    NH_ = 8,
    HD_ = 128,
    KSZ = 4,             // conv kernel size
    I_  = 2816,          // INTER
    M_  = B_ * T_,       // 8192 rows
    H3_ = 3 * H_,        // 3072
};

#define DEV __device__ __forceinline__

typedef __attribute__((ext_vector_type(16))) __bf16        v16bf;
typedef __attribute__((ext_vector_type(8)))  float         v8f;
typedef __attribute__((ext_vector_type(4)))  unsigned int  u32x4;
typedef __attribute__((ext_vector_type(4)))  unsigned int  v4u;
typedef __attribute__((ext_vector_type(4)))  int           v4i;
typedef __attribute__((ext_vector_type(8)))  int           v8i;

struct U32x8 { u32x4 lo, hi; };

#if defined(__has_builtin)
#  if __has_builtin(__builtin_amdgcn_tensor_load_to_lds)
#    define HAS_TDM 1
#  endif
#endif
#ifndef HAS_TDM
#  define HAS_TDM 0
#endif

// ---------------------------------------------------------------------------
// Scalar helpers
// ---------------------------------------------------------------------------
DEV unsigned short f2bf(float f) {
    union { float f; unsigned int u; } c; c.f = f;
    unsigned int u = c.u;
    unsigned int r = u + 0x7fffu + ((u >> 16) & 1u);   // RNE
    return (unsigned short)(r >> 16);
}
DEV float bf2f(unsigned short h) {
    union { unsigned int u; float f; } c; c.u = ((unsigned int)h) << 16;
    return c.f;
}
DEV float sigmoidf_(float x) { return 1.0f / (1.0f + __expf(-x)); }
DEV float siluf_(float x)    { return x * sigmoidf_(x); }
DEV float elu1f_(float x)    { return x > 0.0f ? x + 1.0f : __expf(x); }

// ---------------------------------------------------------------------------
// bf16 WMMA GEMM:  C[M,N] = A[M,K] * W[N,K]^T   (einsum 'mk,nk->mn')
// Tile 128x128, BK=32 (one v_wmma_f32_16x16x32_bf16 K-step), 8 waves/block.
// TDM (tensor_load_to_lds) double-buffered global->LDS staging when available.
// ---------------------------------------------------------------------------
#define BM 128
#define BN 128
#define BKK 32
#define LDST 40   // padded LDS row stride (ushorts): conflict-free, 16B aligned

DEV v16bf load_frag(const unsigned short* p, int g) {
    // 16-bit A-matrix layout: halfs 0..7 = K g*8..g*8+7 ; halfs 8..15 = K 16+g*8..+7
    U32x8 t;
    t.lo = *(const u32x4*)(p + g * 8);
    t.hi = *(const u32x4*)(p + 16 + g * 8);
    return __builtin_bit_cast(v16bf, t);
}

#if HAS_TDM
// Issue one TDM DMA: 128x32 bf16 tile (row stride K elems) -> LDS with 16B row pad
// D# group0: count=1 | lds_addr | global_addr | type=2
// D# group1: data_size=2B, pad_enable, pad_interval=64B, pad_amount=16B,
//            tensor dims huge (tiles never OOB), tile 32x128, dim0 stride = K
DEV void tdm_load_tile_128x32(unsigned lds_byte_off, const unsigned short* gtile, int k_elems) {
    unsigned long long ga = (unsigned long long)gtile;
    v4u g0;
    g0[0] = 1u;                                            // count=1 (valid)
    g0[1] = lds_byte_off;                                  // lds_addr
    g0[2] = (unsigned)ga;                                  // global_addr[31:0]
    g0[3] = (unsigned)((ga >> 32) & 0x01ffffffull) | (2u << 30);  // ga[56:32] | type=2
    v8i g1;
    g1[0] = (int)((1u << 16) | (1u << 20) | (3u << 22) | (3u << 25));
           // data_size=1(2B) | pad_enable | pad_interval=3(16 dw) | pad_amount=3(4 dw)
    g1[1] = (int)0xFFFF0000u;                              // tensor_dim0 lo16 (0x7FFFFFFF)
    g1[2] = (int)0xFFFF7FFFu;                              // tensor_dim0 hi16 | tensor_dim1 lo16
    g1[3] = (int)((32u << 16) | 0x7FFFu);                  // tensor_dim1 hi16 | tile_dim0=32
    g1[4] = 128;                                           // tile_dim1=128, tile_dim2=0
    g1[5] = k_elems;                                       // tensor_dim0_stride[31:0]
    g1[6] = 0;
    g1[7] = 0;
    v4i gz = {};
#if __clang_major__ >= 23
    v8i gz8 = {};
    __builtin_amdgcn_tensor_load_to_lds(g0, g1, gz, gz, gz8, 0);
#else
    __builtin_amdgcn_tensor_load_to_lds(g0, g1, gz, gz, 0);
#endif
}
#endif

__global__ __launch_bounds__(256)
void gemm_bf16_wmma(const unsigned short* __restrict__ A,
                    const unsigned short* __restrict__ W,
                    float* __restrict__ C,
                    int M, int N, int K)
{
    const int tid   = threadIdx.x;
    const int lane  = tid & 31;
    const int wave  = tid >> 5;       // 0..7
    const int waveM = wave >> 1;      // 0..3 : 32-row band
    const int waveN = wave & 1;       // 0..1 : 64-col band
    const int rowB  = blockIdx.y * BM;
    const int colB  = blockIdx.x * BN;

    const int r = lane & 15;          // fragment row (A) / col (D)
    const int g = lane >> 4;          // K-chunk select

    v8f acc[2][4] = {};

#if HAS_TDM
    __shared__ __align__(16) unsigned short As[2][BM][LDST];
    __shared__ __align__(16) unsigned short Ws[2][BN][LDST];

    const unsigned ldsA[2] = { (unsigned)(unsigned long long)&As[0][0][0],
                               (unsigned)(unsigned long long)&As[1][0][0] };
    const unsigned ldsW[2] = { (unsigned)(unsigned long long)&Ws[0][0][0],
                               (unsigned)(unsigned long long)&Ws[1][0][0] };
    const bool issuer = (tid < 32);   // wave 0 issues all DMAs (uniform per wave)
    const int steps = K / BKK;
    const unsigned short* gA = A + (size_t)rowB * K;
    const unsigned short* gW = W + (size_t)colB * K;

    if (issuer) {
        tdm_load_tile_128x32(ldsA[0], gA, K);
        tdm_load_tile_128x32(ldsW[0], gW, K);
    }
    for (int i = 0; i < steps; ++i) {
        const int cur = i & 1;
        if (issuer) __builtin_amdgcn_s_wait_tensorcnt((short)0);  // tiles for buf[cur] done
        __syncthreads();                                          // buf[cur] visible to all
        if (issuer && (i + 1) < steps) {                          // overlap DMA of next tile
            tdm_load_tile_128x32(ldsA[cur ^ 1], gA + (size_t)(i + 1) * BKK, K);
            tdm_load_tile_128x32(ldsW[cur ^ 1], gW + (size_t)(i + 1) * BKK, K);
        }
        v16bf af[2], bfrag[4];
        #pragma unroll
        for (int ii = 0; ii < 2; ++ii)
            af[ii] = load_frag(&As[cur][waveM * 32 + ii * 16 + r][0], g);
        #pragma unroll
        for (int j = 0; j < 4; ++j)
            bfrag[j] = load_frag(&Ws[cur][waveN * 64 + j * 16 + r][0], g);
        #pragma unroll
        for (int ii = 0; ii < 2; ++ii)
            #pragma unroll
            for (int j = 0; j < 4; ++j)
                acc[ii][j] = __builtin_amdgcn_wmma_f32_16x16x32_bf16(
                    false, af[ii], false, bfrag[j],
                    (short)0, acc[ii][j], false, false);
        __syncthreads();   // reads of buf[cur] done before it is DMA'd again (i+2)
    }
#else
    // Fallback: manual staging (global_load_b128 -> ds_store_b128), single buffer
    __shared__ __align__(16) unsigned short As[BM][LDST];
    __shared__ __align__(16) unsigned short Ws[BN][LDST];

    const int ldRow = tid >> 1;       // 0..127
    const int ldCol = (tid & 1) * 16; // 0 or 16 (ushorts)

    for (int kt = 0; kt < K; kt += BKK) {
        const unsigned short* ga = A + (size_t)(rowB + ldRow) * K + kt + ldCol;
        const unsigned short* gw = W + (size_t)(colB + ldRow) * K + kt + ldCol;
        u32x4 a0 = *(const u32x4*)(ga);
        u32x4 a1 = *(const u32x4*)(ga + 8);
        u32x4 w0 = *(const u32x4*)(gw);
        u32x4 w1 = *(const u32x4*)(gw + 8);
        *(u32x4*)&As[ldRow][ldCol]     = a0;
        *(u32x4*)&As[ldRow][ldCol + 8] = a1;
        *(u32x4*)&Ws[ldRow][ldCol]     = w0;
        *(u32x4*)&Ws[ldRow][ldCol + 8] = w1;
        if (kt + BKK < K) {
            __builtin_prefetch(ga + BKK, 0, 0);
            __builtin_prefetch(gw + BKK, 0, 0);
        }
        __syncthreads();
        v16bf af[2], bfrag[4];
        #pragma unroll
        for (int ii = 0; ii < 2; ++ii)
            af[ii] = load_frag(&As[waveM * 32 + ii * 16 + r][0], g);
        #pragma unroll
        for (int j = 0; j < 4; ++j)
            bfrag[j] = load_frag(&Ws[waveN * 64 + j * 16 + r][0], g);
        #pragma unroll
        for (int ii = 0; ii < 2; ++ii)
            #pragma unroll
            for (int j = 0; j < 4; ++j)
                acc[ii][j] = __builtin_amdgcn_wmma_f32_16x16x32_bf16(
                    false, af[ii], false, bfrag[j],
                    (short)0, acc[ii][j], false, false);
        __syncthreads();
    }
#endif

    // ---- store: D layout: VGPR vr, lane<16 -> M=vr, lane>=16 -> M=vr+8, N=lane%16
    #pragma unroll
    for (int i = 0; i < 2; ++i) {
        #pragma unroll
        for (int j = 0; j < 4; ++j) {
            const int row0 = rowB + waveM * 32 + i * 16 + 8 * g;
            const int col  = colB + waveN * 64 + j * 16 + r;
            #pragma unroll
            for (int vr = 0; vr < 8; ++vr)
                C[(size_t)(row0 + vr) * N + col] = acc[i][j][vr];
        }
    }
}

// ---------------------------------------------------------------------------
// Elementwise / normalization / scan kernels
// ---------------------------------------------------------------------------
__global__ void cvt_f32_bf16_kernel(const float* __restrict__ src,
                                    unsigned short* __restrict__ dst, int n) {
    int i = blockIdx.x * blockDim.x + threadIdx.x;
    if (i < n) dst[i] = f2bf(src[i]);
}

// RMSNorm over last dim H, one block per row, bf16 output
__global__ __launch_bounds__(256)
void rmsnorm_bf16_kernel(const float* __restrict__ x, const float* __restrict__ w,
                         unsigned short* __restrict__ out) {
    __shared__ float red[8];
    const int row = blockIdx.x;
    const float* xr = x + (size_t)row * H_;
    float ss = 0.0f;
    for (int c = threadIdx.x; c < H_; c += 256) { float v = xr[c]; ss += v * v; }
    #pragma unroll
    for (int off = 16; off; off >>= 1) ss += __shfl_xor(ss, off, 32);
    const int wv = threadIdx.x >> 5, ln = threadIdx.x & 31;
    if (ln == 0) red[wv] = ss;
    __syncthreads();
    if (wv == 0) {
        float v = (ln < 8) ? red[ln] : 0.0f;
        #pragma unroll
        for (int off = 4; off; off >>= 1) v += __shfl_xor(v, off, 32);
        if (ln == 0) red[0] = v;
    }
    __syncthreads();
    const float scale = rsqrtf(red[0] * (1.0f / H_) + 1e-6f);
    for (int c = threadIdx.x; c < H_; c += 256)
        out[(size_t)row * H_ + c] = f2bf(xr[c] * scale * w[c]);
}

// depthwise causal conv (KS=4) + bias + silu, bf16 in/out
__global__ void conv_silu_kernel(const unsigned short* __restrict__ hb,
                                 const float* __restrict__ cw,
                                 const float* __restrict__ cb,
                                 unsigned short* __restrict__ out) {
    int i = blockIdx.x * blockDim.x + threadIdx.x;        // over M_*H_
    if (i >= M_ * H_) return;
    const int c = i & (H_ - 1);
    const int row = i >> 10;           // /H_
    const int t = row & (T_ - 1);
    float acc = cb[c];
    #pragma unroll
    for (int j = 0; j < KSZ; ++j) {
        const int tt = t - (KSZ - 1) + j;
        if (tt >= 0)
            acc += bf2f(hb[(size_t)(row - (KSZ - 1) + j) * H_ + c]) * cw[c * KSZ + j];
    }
    out[i] = f2bf(siluf_(acc));
}

// decay scans over T: rkv = scan(elu1(k)*v), rks = scan(elu1(k)); one thread per (b,nh,d)
__global__ void attn_scan_kernel(const float* __restrict__ qkv,
                                 const float* __restrict__ decay_param,
                                 float* __restrict__ rkv,
                                 float* __restrict__ rks) {
    int i = blockIdx.x * blockDim.x + threadIdx.x;        // B_*NH_*HD_ = 4096
    if (i >= B_ * NH_ * HD_) return;
    const int d  = i % HD_;
    const int nh = (i / HD_) % NH_;
    const int b  = i / (HD_ * NH_);
    const float dec = sigmoidf_(decay_param[nh]);
    const int col = nh * HD_ + d;
    float skv = 0.0f, sks = 0.0f;
    for (int t = 0; t < T_; ++t) {
        const size_t ro = (size_t)(b * T_ + t) * H3_;
        const float kk = elu1f_(qkv[ro + H_ + col]);
        const float vv = qkv[ro + 2 * H_ + col];
        skv = dec * skv + kk * vv;
        sks = dec * sks + kk;
        const size_t oo = (size_t)(b * T_ + t) * H_ + col;
        rkv[oo] = skv;
        rks[oo] = sks;
    }
}

// o = q' * rkv / max(sum_d q'*rks, 1e-6); one wave per (row, head)
__global__ __launch_bounds__(256)
void attn_o_kernel(const float* __restrict__ qkv,
                   const float* __restrict__ rkv,
                   const float* __restrict__ rks,
                   float* __restrict__ o_f,
                   unsigned short* __restrict__ o_bf) {
    const int lane  = threadIdx.x & 31;
    const int group = blockIdx.x * 8 + (threadIdx.x >> 5); // row*NH + nh
    const int row = group >> 3;
    const int nh  = group & 7;
    const size_t qoff = (size_t)row * H3_ + nh * HD_;
    const size_t roff = (size_t)row * H_  + nh * HD_;
    float qv[4];
    float part = 0.0f;
    #pragma unroll
    for (int k = 0; k < 4; ++k) {
        const int d = lane + 32 * k;
        qv[k] = elu1f_(qkv[qoff + d]);
        part += qv[k] * rks[roff + d];
    }
    #pragma unroll
    for (int off = 16; off; off >>= 1) part += __shfl_xor(part, off, 32);
    const float den = fmaxf(part, 1e-6f);
    #pragma unroll
    for (int k = 0; k < 4; ++k) {
        const int d = lane + 32 * k;
        const float o = qv[k] * rkv[roff + d] / den;
        o_f[roff + d]  = o;
        o_bf[roff + d] = f2bf(o);
    }
}

// g = sigmoid(gbuf + b); o2 = g*o + (1-g)*v_direct  (v_direct from qkv cols 2H..3H)
__global__ void blend_kernel(const float* __restrict__ gbuf,
                             const float* __restrict__ ogb,
                             const float* __restrict__ o_f,
                             const float* __restrict__ qkv,
                             unsigned short* __restrict__ o2_bf) {
    int i = blockIdx.x * blockDim.x + threadIdx.x;        // over M_*H_
    if (i >= M_ * H_) return;
    const int col = i & (H_ - 1);
    const int row = i >> 10;
    const float gg = sigmoidf_(gbuf[i] + ogb[col]);
    const float vd = qkv[(size_t)row * H3_ + 2 * H_ + col];
    o2_bf[i] = f2bf(gg * o_f[i] + (1.0f - gg) * vd);
}

// comb = [local, attn] (bf16)
__global__ void comb_kernel(const float* __restrict__ local_,
                            const float* __restrict__ attn_,
                            unsigned short* __restrict__ comb) {
    int i = blockIdx.x * blockDim.x + threadIdx.x;        // over M_*H_
    if (i >= M_ * H_) return;
    const int col = i & (H_ - 1);
    const int row = i >> 10;
    comb[(size_t)row * 2 * H_ + col]       = f2bf(local_[i]);
    comb[(size_t)row * 2 * H_ + H_ + col]  = f2bf(attn_[i]);
}

// xr = x + s*local + (1-s)*attn,  s = sigmoid(gl)
__global__ void residual_gate_kernel(const float* __restrict__ x,
                                     const float* __restrict__ gl,
                                     const float* __restrict__ local_,
                                     const float* __restrict__ attn_,
                                     float* __restrict__ xr) {
    int i = blockIdx.x * blockDim.x + threadIdx.x;
    if (i >= M_ * H_) return;
    const float s = sigmoidf_(gl[i]);
    xr[i] = x[i] + s * local_[i] + (1.0f - s) * attn_[i];
}

// ff = bf16(silu(u1) * u2)
__global__ void ffn_act_kernel(const float* __restrict__ u1,
                               const float* __restrict__ u2,
                               unsigned short* __restrict__ ff) {
    int i = blockIdx.x * blockDim.x + threadIdx.x;        // over M_*I_
    if (i >= M_ * I_) return;
    ff[i] = f2bf(siluf_(u1[i]) * u2[i]);
}

// out = xr + ffn
__global__ void add_out_kernel(const float* __restrict__ xr,
                               const float* __restrict__ ffn,
                               float* __restrict__ out) {
    int i = blockIdx.x * blockDim.x + threadIdx.x;
    if (i >= M_ * H_) return;
    out[i] = xr[i] + ffn[i];
}

// ---------------------------------------------------------------------------
// Host orchestration
// ---------------------------------------------------------------------------
extern "C" void kernel_launch(void* const* d_in, const int* in_sizes, int n_in,
                              void* d_out, int out_size, void* d_ws, size_t ws_size,
                              hipStream_t stream) {
    (void)in_sizes; (void)n_in; (void)out_size; (void)ws_size;

    const float* x     = (const float*)d_in[0];
    const float* qkv_w = (const float*)d_in[1];
    const float* op_w  = (const float*)d_in[2];
    const float* og_w  = (const float*)d_in[3];
    const float* og_b  = (const float*)d_in[4];
    const float* dec_p = (const float*)d_in[5];
    const float* cv_w  = (const float*)d_in[6];
    const float* cv_b  = (const float*)d_in[7];
    const float* pw_w  = (const float*)d_in[8];
    const float* gt_w  = (const float*)d_in[9];
    const float* w1    = (const float*)d_in[10];
    const float* w2    = (const float*)d_in[11];
    const float* w3    = (const float*)d_in[12];
    const float* n1_w  = (const float*)d_in[13];
    const float* n2_w  = (const float*)d_in[14];
    float* out = (float*)d_out;

    char* ws = (char*)d_ws;
    size_t off = 0;
    auto alloc = [&](size_t bytes) { size_t o = off; off += (bytes + 255) & ~(size_t)255; return o; };

    // persistent bf16 weights
    const size_t wqkv = alloc((size_t)H3_ * H_ * 2);
    const size_t wop  = alloc((size_t)H_ * H_ * 2);
    const size_t wog  = alloc((size_t)H_ * H_ * 2);
    const size_t wpw  = alloc((size_t)H_ * H_ * 2);
    const size_t wgt  = alloc((size_t)H_ * 2 * H_ * 2);
    const size_t ww1  = alloc((size_t)I_ * H_ * 2);
    const size_t ww2  = alloc((size_t)I_ * H_ * 2);
    const size_t ww3  = alloc((size_t)H_ * I_ * 2);
    // persistent-ish activations
    const size_t hb    = alloc((size_t)M_ * H_ * 2);   // rmsnorm1 bf16
    const size_t local = alloc((size_t)M_ * H_ * 4);
    const size_t attnb = alloc((size_t)M_ * H_ * 4);
    const size_t xr    = alloc((size_t)M_ * H_ * 4);
    const size_t h2b   = alloc((size_t)M_ * H_ * 2);
    // big aliased scratch region
    const size_t big   = alloc((size_t)285212672);
    // phase A (attention)
    const size_t qkvb = big;                                       // M*3072 f32
    const size_t rkv  = big + (size_t)M_ * H3_ * 4;
    const size_t rks  = rkv + (size_t)M_ * H_ * 4;
    const size_t o_f  = rks + (size_t)M_ * H_ * 4;
    const size_t gbuf = o_f + (size_t)M_ * H_ * 4;
    const size_t o_bf = gbuf + (size_t)M_ * H_ * 4;
    const size_t o2b  = o_bf + (size_t)M_ * H_ * 2;
    const size_t cvb  = o2b + (size_t)M_ * H_ * 2;
    // phase B (gate fusion) — aliases dead qkv
    const size_t comb = big;
    const size_t gl   = big + (size_t)M_ * 2 * H_ * 2;
    // phase C (FFN) — aliases everything above
    const size_t u1   = big;
    const size_t u2   = u1 + (size_t)M_ * I_ * 4;
    const size_t ffb  = u2 + (size_t)M_ * I_ * 4;
    const size_t ffnb = ffb + (size_t)M_ * I_ * 2;

    auto cvt = [&](const float* src, size_t dst, int n) {
        cvt_f32_bf16_kernel<<<(n + 255) / 256, 256, 0, stream>>>(
            src, (unsigned short*)(ws + dst), n);
    };
    auto gemm = [&](size_t a, size_t w, size_t c, int M, int N, int K) {
        dim3 grid((unsigned)(N / BN), (unsigned)(M / BM));
        gemm_bf16_wmma<<<grid, 256, 0, stream>>>(
            (const unsigned short*)(ws + a), (const unsigned short*)(ws + w),
            (float*)(ws + c), M, N, K);
    };

    // --- weight conversion (every call; deterministic) ---
    cvt(qkv_w, wqkv, H3_ * H_);
    cvt(op_w,  wop,  H_ * H_);
    cvt(og_w,  wog,  H_ * H_);
    cvt(pw_w,  wpw,  H_ * H_);
    cvt(gt_w,  wgt,  H_ * 2 * H_);
    cvt(w1,    ww1,  I_ * H_);
    cvt(w2,    ww2,  I_ * H_);
    cvt(w3,    ww3,  H_ * I_);

    const int EW = (M_ * H_) / 256;   // 32768 blocks for M*H elementwise

    // 1) h = rmsnorm(x, norm1_w) -> bf16
    rmsnorm_bf16_kernel<<<M_, 256, 0, stream>>>(x, n1_w, (unsigned short*)(ws + hb));
    // 2) depthwise conv + silu
    conv_silu_kernel<<<EW, 256, 0, stream>>>((const unsigned short*)(ws + hb), cv_w, cv_b,
                                             (unsigned short*)(ws + cvb));
    // 3) qkv = h @ qkv_w^T
    gemm(hb, wqkv, qkvb, M_, H3_, H_);
    // 4) local = silu(conv) @ pw_w^T
    gemm(cvb, wpw, local, M_, H_, H_);
    // 5) decay scans
    attn_scan_kernel<<<(B_ * NH_ * HD_ + 255) / 256, 256, 0, stream>>>(
        (const float*)(ws + qkvb), dec_p, (float*)(ws + rkv), (float*)(ws + rks));
    // 6) o = q'*rkv/den
    attn_o_kernel<<<(M_ * NH_) / 8, 256, 0, stream>>>(
        (const float*)(ws + qkvb), (const float*)(ws + rkv), (const float*)(ws + rks),
        (float*)(ws + o_f), (unsigned short*)(ws + o_bf));
    // 7) gate = o @ out_gate_w^T
    gemm(o_bf, wog, gbuf, M_, H_, H_);
    // 8) o2 = g*o + (1-g)*v_direct
    blend_kernel<<<EW, 256, 0, stream>>>((const float*)(ws + gbuf), og_b,
                                         (const float*)(ws + o_f), (const float*)(ws + qkvb),
                                         (unsigned short*)(ws + o2b));
    // 9) attn = o2 @ out_proj_w^T
    gemm(o2b, wop, attnb, M_, H_, H_);
    // 10) comb = [local, attn]
    comb_kernel<<<EW, 256, 0, stream>>>((const float*)(ws + local), (const float*)(ws + attnb),
                                        (unsigned short*)(ws + comb));
    // 11) gl = comb @ gate_w^T
    gemm(comb, wgt, gl, M_, H_, 2 * H_);
    // 12) xr = x + s*local + (1-s)*attn
    residual_gate_kernel<<<EW, 256, 0, stream>>>(x, (const float*)(ws + gl),
                                                 (const float*)(ws + local),
                                                 (const float*)(ws + attnb),
                                                 (float*)(ws + xr));
    // 13) h2 = rmsnorm(xr, norm2_w)
    rmsnorm_bf16_kernel<<<M_, 256, 0, stream>>>((const float*)(ws + xr), n2_w,
                                                (unsigned short*)(ws + h2b));
    // 14) u1 = h2 @ w1^T, u2 = h2 @ w2^T
    gemm(h2b, ww1, u1, M_, I_, H_);
    gemm(h2b, ww2, u2, M_, I_, H_);
    // 15) ff = silu(u1)*u2
    ffn_act_kernel<<<(M_ * I_) / 256, 256, 0, stream>>>((const float*)(ws + u1),
                                                        (const float*)(ws + u2),
                                                        (unsigned short*)(ws + ffb));
    // 16) ffn = ff @ w3^T
    gemm(ffb, ww3, ffnb, M_, H_, I_);
    // 17) out = xr + ffn
    add_out_kernel<<<EW, 256, 0, stream>>>((const float*)(ws + xr), (const float*)(ws + ffnb), out);
}
